// RoPEMultiHeadSelfAttention_32066225832665
// MI455X (gfx1250) — matrix operand order repaired and checked by reading, not compiled, not run on hardware
//
#include <hip/hip_runtime.h>
#include <hip/hip_bf16.h>

typedef __bf16 bf16;
typedef __attribute__((ext_vector_type(16))) __bf16 v16bf;
typedef __attribute__((ext_vector_type(8)))  __bf16 v8bf;
typedef __attribute__((ext_vector_type(8)))  float  v8f;

#define B_   2
#define S_   2048
#define E_   768
#define H_   12
#define D_   64
#define KDIM 768   // reduction dim of all projection GEMMs
#define MROWS 4096 // B_*S_

static __device__ __forceinline__ v16bf cat8(v8bf lo, v8bf hi) {
  return __builtin_shufflevector(lo, hi, 0,1,2,3,4,5,6,7,8,9,10,11,12,13,14,15);
}
static __device__ __forceinline__ v8f wmma_bf16(v16bf a, v16bf b, v8f c) {
  return __builtin_amdgcn_wmma_f32_16x16x32_bf16(false, a, false, b, (short)0, c, false, false);
}

// ---------------------------------------------------------------- converts
__global__ void f32_to_bf16_kernel(const float* __restrict__ a, bf16* __restrict__ b, int n) {
  int i = blockIdx.x * blockDim.x + threadIdx.x;
  if (i < n) b[i] = (bf16)a[i];
}
// W is [K x N] row-major f32; Wt is [N x K] bf16 (so B-fragments load contiguous K)
__global__ void w_transpose_bf16_kernel(const float* __restrict__ W, bf16* __restrict__ Wt,
                                        int K, int N) {
  int idx = blockIdx.x * blockDim.x + threadIdx.x;
  if (idx >= K * N) return;
  int k = idx / N, n = idx % N;
  Wt[(size_t)n * K + k] = (bf16)W[idx];
}

// ---------------------------------------------------------------- GEMM + epilogues
// MODE 0: QK projection -> bias,scale,RoPE, write bf16 [b,h,s,d] into o1 (cols<768) / o2
// MODE 1: V  projection -> bias,scale, write bf16 transposed [b,h,d,s] into ovt
// MODE 2: out projection -> bias,scale, write f32 [m, col] into of
template <int MODE>
__global__ void proj_gemm_kernel(const bf16* __restrict__ A,   // [4096 x 768] bf16
                                 const bf16* __restrict__ Wt,  // [N x 768] bf16
                                 const float* __restrict__ bias,
                                 const float* __restrict__ scl,
                                 bf16* __restrict__ o1, bf16* __restrict__ o2,
                                 bf16* __restrict__ ovt, float* __restrict__ of,
                                 int N) {
  const int ntiles = N >> 6;                       // 64-wide N tiles per wave
  int wid = blockIdx.x * (blockDim.x >> 5) + (threadIdx.x >> 5);
  if (wid >= (MROWS / 16) * ntiles) return;
  int mt = wid / ntiles, nt = wid % ntiles;
  int m0 = mt * 16, n0 = nt * 64;
  int lane = threadIdx.x & 31, ln = lane & 15, hf = lane >> 4;
  int arow = m0 + ln;

  v8f acc[4] = {};
  for (int k0 = 0; k0 < KDIM; k0 += 32) {
    const bf16* ap = A + (size_t)arow * KDIM + k0 + hf * 16;
    v16bf a = cat8(*(const v8bf*)ap, *(const v8bf*)(ap + 8));
#pragma unroll
    for (int t = 0; t < 4; ++t) {
      const bf16* bp = Wt + (size_t)(n0 + t * 16 + ln) * KDIM + k0 + hf * 16;
      acc[t] = wmma_bf16(a, *(const v16bf*)bp, acc[t]);
    }
  }

#pragma unroll
  for (int t = 0; t < 4; ++t) {
    int col = n0 + t * 16 + ln;
    float bv = bias[col], sv = scl[col];
#pragma unroll
    for (int r = 0; r < 8; ++r) {
      int m = m0 + r + 8 * hf;               // token row
      float v = (acc[t][r] + bv) * sv;
      if (MODE == 2) {
        of[(size_t)m * E_ + col] = v;
      } else if (MODE == 1) {
        int h = col / D_, d = col % D_;
        int b = m >> 11, s = m & (S_ - 1);
        ovt[((size_t)(b * H_ + h) * D_ + d) * S_ + s] = (bf16)v;
      } else {
        int cm = col & (E_ - 1);
        int h = cm / D_, d = cm % D_, j = d >> 1;
        int b = m >> 11, s = m & (S_ - 1);
        // theta = 10000^(-2j/64) = exp(j * -ln(10000)/32)
        float theta = __expf(-0.2878231366f * (float)j);
        float ang = (float)s * theta;
        float c = __cosf(ang), sn = __sinf(ang);
        float p = __shfl_xor(v, 1, 32);      // pair partner (d even<->odd)
        float rv = (ln & 1) ? (p * sn + v * c) : (v * c - p * sn);
        bf16* dst = (col < E_) ? o1 : o2;
        dst[((size_t)(b * H_ + h) * S_ + s) * D_ + d] = (bf16)rv;
      }
    }
  }
}

// ---------------------------------------------------------------- flash differential attention
// S^T formulation: one wave = 16 queries (columns) of one (b,h).
// Per stream st: S^T = K*Q^T (rows=keys), softmax stats are PER-LANE (one query per lane,
// one shfl_xor(16) to reach partner half), ctx^T = V^T * P^T accumulated transposed.
__global__ void __launch_bounds__(128, 1)
attn_kernel(const bf16* __restrict__ q1, const bf16* __restrict__ q2,
            const bf16* __restrict__ k1, const bf16* __restrict__ k2,
            const bf16* __restrict__ vt, const float* __restrict__ lambda_p,
            bf16* __restrict__ ctx) {
  __shared__ __align__(64) bf16 plds[4][16 * 32];   // per-wave P^T re-layout: [query][key]
  int wib = threadIdx.x >> 5;
  int wid = blockIdx.x * 4 + wib;
  if (wid >= (B_ * H_) * (S_ / 16)) return;
  int qt = wid & (S_ / 16 - 1);
  int bh = wid >> 7;                                 // b*H + h
  int lane = threadIdx.x & 31, ln = lane & 15, hf = lane >> 4;
  float lam = 1.f / (1.f + __expf(-lambda_p[0]));
  size_t base = (size_t)bh * S_ * D_;
  const bf16* Vb = vt + base;                        // [d][s] for this (b,h)
  int q0 = qt * 16;
  bf16* pl = plds[wib];

  v8f ctx1[4];                                       // stream-0 result, then combined

#pragma unroll 1
  for (int st = 0; st < 2; ++st) {
    // ternary select (not a pointer array) so infer-address-spaces keeps GLOBAL loads
    const bf16* Qp = ((st == 0) ? q1 : q2) + base;
    const bf16* Kp = ((st == 0) ? k1 : k2) + base;

    // Q^T B-fragments: lane = query column, contiguous dims
    v16bf qb[2];
#pragma unroll
    for (int kk = 0; kk < 2; ++kk)
      qb[kk] = *(const v16bf*)(Qp + (size_t)(q0 + ln) * D_ + kk * 32 + hf * 16);

    v8f acc[4] = {};
    float m = -3.0e38f, l = 0.f;

    for (int kc = 0; kc < S_; kc += 32) {
      if (kc + 32 < S_) {
        // next K chunk: 32 lanes cover the 32 key rows (128B each = 1 line)
        __builtin_prefetch(Kp + (size_t)(kc + 32 + lane) * D_, 0, 3);
        __builtin_prefetch(Vb + (size_t)lane * S_ + kc + 32, 0, 3);
        __builtin_prefetch(Vb + (size_t)(32 + lane) * S_ + kc + 32, 0, 3);
      }
      // S^T tiles: rows = keys (t*16 + r + 8*hf), cols = queries (ln)
      v8f sc[2] = {};
#pragma unroll
      for (int t = 0; t < 2; ++t)
#pragma unroll
        for (int kk = 0; kk < 2; ++kk) {
          const bf16* ap = Kp + (size_t)(kc + t * 16 + ln) * D_ + kk * 32 + hf * 16;
          v16bf af = cat8(*(const v8bf*)ap, *(const v8bf*)(ap + 8));
          sc[t] = wmma_bf16(af, qb[kk], sc[t]);
        }
      // per-lane online softmax over this lane's 16 keys; partner half via one shuffle
      float mx = -3.0e38f;
#pragma unroll
      for (int t = 0; t < 2; ++t)
#pragma unroll
        for (int r = 0; r < 8; ++r) mx = fmaxf(mx, sc[t][r]);
      mx *= 0.125f;
      mx = fmaxf(mx, __shfl_xor(mx, 16, 32));
      float mn = fmaxf(m, mx);
      float corr = __expf(m - mn);
      float rs = 0.f;
#pragma unroll
      for (int t = 0; t < 2; ++t)
#pragma unroll
        for (int r = 0; r < 8; ++r) {
          float p = __expf(sc[t][r] * 0.125f - mn);
          rs += p;
          pl[ln * 32 + t * 16 + r + 8 * hf] = (bf16)p;   // [query][key]
        }
      rs += __shfl_xor(rs, 16, 32);
      l = l * corr + rs;
      m = mn;
#pragma unroll
      for (int mt = 0; mt < 4; ++mt)
#pragma unroll
        for (int r = 0; r < 8; ++r) acc[mt][r] *= corr;  // uniform per lane
      // P^T B-fragment: lane = query, 16 contiguous keys (in-wave LDS ordering)
      v16bf pb = cat8(*(const v8bf*)(pl + ln * 32 + hf * 16),
                      *(const v8bf*)(pl + ln * 32 + hf * 16 + 8));
#pragma unroll
      for (int mt = 0; mt < 4; ++mt) {
        const bf16* vp = Vb + (size_t)(mt * 16 + ln) * S_ + kc + hf * 16;
        v16bf va = cat8(*(const v8bf*)vp, *(const v8bf*)(vp + 8));
        acc[mt] = wmma_bf16(va, pb, acc[mt]);            // ctx^T += V^T P^T
      }
    }
    float invl = 1.f / l;
    if (st == 0) {
#pragma unroll
      for (int mt = 0; mt < 4; ++mt)
#pragma unroll
        for (int r = 0; r < 8; ++r) ctx1[mt][r] = acc[mt][r] * invl;
    } else {
      float li = lam * invl;
#pragma unroll
      for (int mt = 0; mt < 4; ++mt)
#pragma unroll
        for (int r = 0; r < 8; ++r) ctx1[mt][r] -= acc[mt][r] * li;
    }
  }

  // LayerNorm over D=64 (rows of ctx^T): in-lane sum of 32 dims + partner half
  float sum = 0.f;
#pragma unroll
  for (int mt = 0; mt < 4; ++mt)
#pragma unroll
    for (int r = 0; r < 8; ++r) sum += ctx1[mt][r];
  sum += __shfl_xor(sum, 16, 32);
  float mean = sum * (1.f / 64.f);
  float vs = 0.f;
#pragma unroll
  for (int mt = 0; mt < 4; ++mt)
#pragma unroll
    for (int r = 0; r < 8; ++r) { float d = ctx1[mt][r] - mean; vs += d * d; }
  vs += __shfl_xor(vs, 16, 32);
  float rstd = rsqrtf(vs * (1.f / 64.f) + 1e-5f);

  int bb = bh / H_, hh = bh % H_;
  bf16* op = ctx + (size_t)(bb * S_ + q0 + ln) * E_ + hh * D_;   // lane = query token
#pragma unroll
  for (int mt = 0; mt < 4; ++mt)
#pragma unroll
    for (int r = 0; r < 8; ++r)
      op[mt * 16 + r + 8 * hf] = (bf16)((ctx1[mt][r] - mean) * rstd * 0.5f);
}

// ---------------------------------------------------------------- host
extern "C" void kernel_launch(void* const* d_in, const int* in_sizes, int n_in,
                              void* d_out, int out_size, void* d_ws, size_t ws_size,
                              hipStream_t stream) {
  const float* x  = (const float*)d_in[0];
  const float* Wq = (const float*)d_in[1];  const float* bq = (const float*)d_in[2];
  const float* Wk = (const float*)d_in[3];  const float* bk = (const float*)d_in[4];
  const float* Wv = (const float*)d_in[5];  const float* bv = (const float*)d_in[6];
  const float* Wo = (const float*)d_in[7];  const float* bo = (const float*)d_in[8];
  const float* qs = (const float*)d_in[9];  const float* ks = (const float*)d_in[10];
  const float* vs = (const float*)d_in[11]; const float* os = (const float*)d_in[12];
  const float* lp = (const float*)d_in[13];

  char* ws = (char*)d_ws;
  size_t off = 0;
  auto alloc = [&](size_t bytes) -> void* {
    void* p = ws + off; off += (bytes + 255) & ~(size_t)255; return p;
  };
  bf16* xbf  = (bf16*)alloc((size_t)MROWS * E_ * 2);
  bf16* wqT  = (bf16*)alloc((size_t)2 * E_ * E_ * 2);
  bf16* wkT  = (bf16*)alloc((size_t)2 * E_ * E_ * 2);
  bf16* wvT  = (bf16*)alloc((size_t)E_ * E_ * 2);
  bf16* woT  = (bf16*)alloc((size_t)E_ * E_ * 2);
  bf16* q1b  = (bf16*)alloc((size_t)MROWS * E_ * 2);
  bf16* q2b  = (bf16*)alloc((size_t)MROWS * E_ * 2);
  bf16* k1b  = (bf16*)alloc((size_t)MROWS * E_ * 2);
  bf16* k2b  = (bf16*)alloc((size_t)MROWS * E_ * 2);
  bf16* vtb  = (bf16*)alloc((size_t)MROWS * E_ * 2);
  bf16* ctxb = (bf16*)alloc((size_t)MROWS * E_ * 2);

  // 1) converts
  int nx = MROWS * E_;
  f32_to_bf16_kernel<<<(nx + 255) / 256, 256, 0, stream>>>(x, xbf, nx);
  w_transpose_bf16_kernel<<<(E_ * 2 * E_ + 255) / 256, 256, 0, stream>>>(Wq, wqT, E_, 2 * E_);
  w_transpose_bf16_kernel<<<(E_ * 2 * E_ + 255) / 256, 256, 0, stream>>>(Wk, wkT, E_, 2 * E_);
  w_transpose_bf16_kernel<<<(E_ * E_ + 255) / 256, 256, 0, stream>>>(Wv, wvT, E_, E_);
  w_transpose_bf16_kernel<<<(E_ * E_ + 255) / 256, 256, 0, stream>>>(Wo, woT, E_, E_);

  // 2) projections (wave = 16x64 tile; 4 waves / 128-thread block)
  int wavesQK = (MROWS / 16) * (2 * E_ / 64);   // 6144
  int wavesV  = (MROWS / 16) * (E_ / 64);       // 3072
  proj_gemm_kernel<0><<<wavesQK / 4, 128, 0, stream>>>(xbf, wqT, bq, qs, q1b, q2b, nullptr, nullptr, 2 * E_);
  proj_gemm_kernel<0><<<wavesQK / 4, 128, 0, stream>>>(xbf, wkT, bk, ks, k1b, k2b, nullptr, nullptr, 2 * E_);
  proj_gemm_kernel<1><<<wavesV  / 4, 128, 0, stream>>>(xbf, wvT, bv, vs, nullptr, nullptr, vtb, nullptr, E_);

  // 3) differential flash attention (wave = 16 queries; 3072 waves)
  attn_kernel<<<(B_ * H_ * (S_ / 16)) / 4, 128, 0, stream>>>(q1b, q2b, k1b, k2b, vtb, lp, ctxb);

  // 4) output projection -> f32
  proj_gemm_kernel<2><<<wavesV / 4, 128, 0, stream>>>(ctxb, woT, bo, os, nullptr, nullptr, nullptr,
                                                      (float*)d_out, E_);
}
